// RecurrentBlock_27401891349122
// MI455X (gfx1250) — matrix operand order repaired and checked by reading, not compiled, not run on hardware
//
#include <hip/hip_runtime.h>

// ---------------------------------------------------------------------------
// Problem constants (match the reference)
// ---------------------------------------------------------------------------
#define BB 4
#define TT 2048
#define DD 2048
#define HH 8
#define TW 4
#define DH 256            // DD / HH
#define MT (BB * TT)      // 8192 rows for all GEMMs
#define NC 32             // scan chunks
#define TC (TT / NC)      // 64 steps per chunk

// WMMA fragment types (gfx1250, wave32)
typedef __attribute__((ext_vector_type(16))) __bf16 v16bf;
typedef __attribute__((ext_vector_type(8)))  float  v8f;

union FragBF {
    v16bf v;
    uint4 q[2];
};

__device__ __forceinline__ unsigned short f2bf(float f) {
    unsigned int u = __float_as_uint(f);
    unsigned int r = u + 0x7FFFu + ((u >> 16) & 1u);   // round-to-nearest-even
    return (unsigned short)(r >> 16);
}

__device__ __forceinline__ float gelu_tanh(float v) {
    return 0.5f * v * (1.f + tanhf(0.7978845608028654f * (v + 0.044715f * v * v * v)));
}

// CDNA5 async global->LDS copy (ASYNCcnt-tracked), per CDNA5 ISA
__device__ __forceinline__ void async_copy_b128(unsigned lds_off, const void* gaddr) {
    asm volatile("global_load_async_to_lds_b128 %0, %1, off"
                 :: "v"(lds_off), "v"(gaddr)
                 : "memory");
}
__device__ __forceinline__ void wait_async0() {
    asm volatile("s_wait_asynccnt 0x0" ::: "memory");
}

// ---------------------------------------------------------------------------
// f32 -> bf16 conversion
// ---------------------------------------------------------------------------
__global__ __launch_bounds__(256) void k_f32_to_bf16(const float* __restrict__ in,
                                                     unsigned short* __restrict__ out,
                                                     int n) {
    int i = blockIdx.x * 256 + threadIdx.x;
    if (i < n) out[i] = f2bf(in[i]);
}

// Gate weights [H][i][j] -> bf16 [H][j][i] so every GEMM uses the [N,K] path.
__global__ __launch_bounds__(256) void k_transpose_gate(const float* __restrict__ in,
                                                        unsigned short* __restrict__ out) {
    int idx = blockIdx.x * 256 + threadIdx.x;
    if (idx >= HH * DH * DH) return;
    int h = idx / (DH * DH);
    int r = idx % (DH * DH);
    int i = r / DH, j = r % DH;
    out[h * DH * DH + j * DH + i] = f2bf(in[idx]);
}

// ---------------------------------------------------------------------------
// WMMA bf16 GEMM:  C[m, cOff+n] = act( sum_k A[m, aOff+k] * W[n, k] + bias[cOff+n] )
// W is [N,K] row-major. Block tile 128x128, 8 waves (4 M x 2 N), wave tile
// 32x64 (2x4 WMMA 16x16x32 tiles -> 8 WMMA per k-step). Double-buffered LDS
// filled with ASYNCcnt-tracked async b128 copies overlapping the WMMAs.
// ---------------------------------------------------------------------------
template <int ACT>
__global__ __launch_bounds__(256) void k_gemm_bf16(
    const unsigned short* __restrict__ A,   // bf16 bits [M, lda]
    const unsigned short* __restrict__ Wm,  // bf16 bits [N, ldb]
    const float* __restrict__ bias,         // indexed by global output column
    float* __restrict__ C,                  // [M, ldc]
    int K, int lda, int ldb, int ldc,
    int aColStride, long wBatchStride, int cColStride) {

    __shared__ __align__(16) unsigned short As[2][128][40];  // [m][k], 80B rows
    __shared__ __align__(16) unsigned short Bs[2][128][40];  // [n][k], 80B rows

    const int tid  = threadIdx.x;
    const int lane = tid & 31;
    const int wave = tid >> 5;
    const int wm   = wave >> 1;   // 0..3  -> wave M tile of 32
    const int wn   = wave & 1;    // 0..1  -> wave N tile of 64

    const int  m0   = blockIdx.y * 128;
    const int  n0   = blockIdx.x * 128;
    const int  z    = blockIdx.z;
    const int  aOff = z * aColStride;
    const long wOff = (long)z * wBatchStride;
    const int  cOff = z * cColStride;

    // staging assignments (128-bit chunks): 512 chunks per 128x32 tile
    const int r0 = tid >> 2;               // rows 0..63
    const int r1 = r0 + 64;                // rows 64..127
    const int cc = (tid & 3) << 3;         // k-chunk 0/8/16/24

    v8f acc[2][4];
    #pragma unroll
    for (int mi = 0; mi < 2; ++mi)
        #pragma unroll
        for (int ni = 0; ni < 4; ++ni)
            acc[mi][ni] = (v8f){0.f, 0.f, 0.f, 0.f, 0.f, 0.f, 0.f, 0.f};

    // ISA 16-bit A layout: lanes 0-15 M=lane, K halves {0..7,16..23}; lanes 16-31 +8 on K.
    const int arow  = wm * 32 + (lane & 15);
    const int akoff = (lane >> 4) << 3;          // 0 or 8
    // 16-bit B layout (matches A): lanes 0-15 N=lane, K halves {0..7,16..23}; lanes 16-31 +8 on K.
    const int bcol  = wn * 64 + (lane & 15);
    const int bkoff = (lane >> 4) << 3;          // 0 or 8

    auto stage = [&](int kt, int buf) {
        const int k0 = kt << 5;
        async_copy_b128((unsigned)(unsigned long long)&As[buf][r0][cc],
                        &A[(size_t)(m0 + r0) * lda + aOff + k0 + cc]);
        async_copy_b128((unsigned)(unsigned long long)&As[buf][r1][cc],
                        &A[(size_t)(m0 + r1) * lda + aOff + k0 + cc]);
        async_copy_b128((unsigned)(unsigned long long)&Bs[buf][r0][cc],
                        &Wm[wOff + (size_t)(n0 + r0) * ldb + k0 + cc]);
        async_copy_b128((unsigned)(unsigned long long)&Bs[buf][r1][cc],
                        &Wm[wOff + (size_t)(n0 + r1) * ldb + k0 + cc]);
    };

    const int nk = K >> 5;
    stage(0, 0);
    wait_async0();
    __syncthreads();

    for (int kt = 0; kt < nk; ++kt) {
        const int buf = kt & 1;
        if (kt + 1 < nk) stage(kt + 1, buf ^ 1);   // overlap with compute below

        FragBF a[2], b[4];
        #pragma unroll
        for (int mi = 0; mi < 2; ++mi) {
            a[mi].q[0] = *(const uint4*)&As[buf][arow + mi * 16][akoff];
            a[mi].q[1] = *(const uint4*)&As[buf][arow + mi * 16][akoff + 16];
        }
        #pragma unroll
        for (int ni = 0; ni < 4; ++ni) {
            b[ni].q[0] = *(const uint4*)&Bs[buf][bcol + ni * 16][bkoff];
            b[ni].q[1] = *(const uint4*)&Bs[buf][bcol + ni * 16][bkoff + 16];
        }

        #pragma unroll
        for (int mi = 0; mi < 2; ++mi)
            #pragma unroll
            for (int ni = 0; ni < 4; ++ni)
                acc[mi][ni] = __builtin_amdgcn_wmma_f32_16x16x32_bf16(
                    false, a[mi].v, false, b[ni].v, (short)0, acc[mi][ni],
                    false, false);

        if (kt + 1 < nk) wait_async0();            // own async copies done
        __syncthreads();                           // everyone done reading+writing
    }

    // Epilogue: C/D layout — VGPR r: lanes 0-15 (M=r, N=lane), lanes 16-31 (M=r+8, N=lane-16)
    #pragma unroll
    for (int mi = 0; mi < 2; ++mi) {
        #pragma unroll
        for (int ni = 0; ni < 4; ++ni) {
            const int col   = cOff + n0 + wn * 64 + ni * 16 + (lane & 15);
            const float bv  = bias[col];
            const int rbase = m0 + wm * 32 + mi * 16 + ((lane >> 4) << 3);
            #pragma unroll
            for (int r = 0; r < 8; ++r) {
                float v = acc[mi][ni][r] + bv;
                if (ACT == 1) v = gelu_tanh(v);
                C[(size_t)(rbase + r) * ldc + col] = v;
            }
        }
    }
}

// ---------------------------------------------------------------------------
// Depthwise causal conv over T (TW taps), segment-masked; writes f32 + bf16
// ---------------------------------------------------------------------------
__global__ __launch_bounds__(256) void k_conv(const float* __restrict__ xp,
                                              const float* __restrict__ cw,
                                              const float* __restrict__ cb,
                                              const int* __restrict__ segpos,
                                              float* __restrict__ xc,
                                              unsigned short* __restrict__ xcb) {
    int idx = blockIdx.x * 256 + threadIdx.x;
    if (idx >= MT * DD) return;
    int d  = idx % DD;
    int mt = idx / DD;       // b*T + t
    int t  = mt % TT;
    int pos = segpos[mt];
    float acc = cb[d];
    #pragma unroll
    for (int i = 0; i < TW; ++i) {
        int shift = TW - 1 - i;
        if (t - shift >= 0 && pos >= shift)
            acc += xp[idx - shift * DD] * cw[i * DD + d];
    }
    xc[idx]  = acc;
    xcb[idx] = f2bf(acc);
}

// ---------------------------------------------------------------------------
// RG-LRU gate prep (biases already applied by GEMM epilogue):
//   gx buffer <- normed (in-place), ga buffer <- a_scan
// ---------------------------------------------------------------------------
__global__ __launch_bounds__(256) void k_prep(float* __restrict__ gx,
                                              float* __restrict__ ga,
                                              const float* __restrict__ xc,
                                              const float* __restrict__ ap,
                                              const int* __restrict__ segpos) {
    int idx = blockIdx.x * 256 + threadIdx.x;
    if (idx >= MT * DD) return;
    int d  = idx % DD;
    int mt = idx / DD;
    float gate_x = 1.f / (1.f + expf(-gx[idx]));
    float gate_a = 1.f / (1.f + expf(-ga[idx]));
    float apv = ap[d];
    float sp  = (apv > 20.f) ? apv : log1pf(expf(apv));   // softplus
    float log_a = -8.f * gate_a * sp;
    float a = expf(log_a);
    bool reset = (segpos[mt] == 0);
    float mult = sqrtf(fmaxf(1.f - expf(2.f * log_a), 0.f));
    if (reset) mult = 1.f;
    gx[idx] = xc[idx] * gate_x * mult;   // normed
    ga[idx] = reset ? 0.f : a;           // a_scan
}

// ---------------------------------------------------------------------------
// Chunked parallel scan (3 passes): h[t] = a[t]*h[t-1] + xn[t]
// ---------------------------------------------------------------------------
__global__ __launch_bounds__(256) void k_scan_partial(const float* __restrict__ a,
                                                      const float* __restrict__ xn,
                                                      float* __restrict__ P,
                                                      float* __restrict__ S) {
    int idx = blockIdx.x * 256 + threadIdx.x;       // over B*NC*D
    if (idx >= BB * NC * DD) return;
    int d     = idx % DD;
    int chunk = (idx / DD) % NC;
    int b     = idx / (DD * NC);
    size_t base = ((size_t)b * TT + (size_t)chunk * TC) * DD + d;
    float p = 1.f, s = 0.f;
    for (int t = 0; t < TC; ++t) {
        size_t i = base + (size_t)t * DD;
        float av = a[i];
        s = av * s + xn[i];
        p *= av;
    }
    P[idx] = p;
    S[idx] = s;
}

__global__ __launch_bounds__(256) void k_scan_carry(const float* __restrict__ P,
                                                    const float* __restrict__ S,
                                                    float* __restrict__ carry) {
    int ch = blockIdx.x * 256 + threadIdx.x;        // over B*D
    if (ch >= BB * DD) return;
    int b = ch / DD, d = ch % DD;
    float h = 0.f;
    for (int c = 0; c < NC; ++c) {
        int idx = (b * NC + c) * DD + d;
        carry[idx] = h;
        h = P[idx] * h + S[idx];
    }
}

__global__ __launch_bounds__(256) void k_scan_final(const float* __restrict__ a,
                                                    const float* __restrict__ xn,
                                                    const float* __restrict__ y,
                                                    const float* __restrict__ carry,
                                                    unsigned short* __restrict__ zb) {
    int idx = blockIdx.x * 256 + threadIdx.x;       // over B*NC*D
    if (idx >= BB * NC * DD) return;
    int d     = idx % DD;
    int chunk = (idx / DD) % NC;
    int b     = idx / (DD * NC);
    size_t base = ((size_t)b * TT + (size_t)chunk * TC) * DD + d;
    float h = carry[idx];
    for (int t = 0; t < TC; ++t) {
        size_t i = base + (size_t)t * DD;
        h = a[i] * h + xn[i];
        zb[i] = f2bf(h * y[i]);
    }
}

// ---------------------------------------------------------------------------
// Launcher
// ---------------------------------------------------------------------------
extern "C" void kernel_launch(void* const* d_in, const int* in_sizes, int n_in,
                              void* d_out, int out_size, void* d_ws, size_t ws_size,
                              hipStream_t stream) {
    (void)in_sizes; (void)n_in; (void)out_size; (void)ws_size;

    const float* x       = (const float*)d_in[0];
    const int*   segpos  = (const int*)d_in[1];
    const float* Wy      = (const float*)d_in[2];
    const float* by      = (const float*)d_in[3];
    const float* Wx      = (const float*)d_in[4];
    const float* bx      = (const float*)d_in[5];
    const float* conv_w  = (const float*)d_in[6];
    const float* conv_b  = (const float*)d_in[7];
    const float* gx_w    = (const float*)d_in[8];
    const float* gx_b    = (const float*)d_in[9];
    const float* ga_w    = (const float*)d_in[10];
    const float* ga_b    = (const float*)d_in[11];
    const float* a_param = (const float*)d_in[12];
    const float* Wout    = (const float*)d_in[13];
    const float* bout    = (const float*)d_in[14];
    float* out = (float*)d_out;

    // Workspace layout (256B aligned slabs)
    char* w = (char*)d_ws;
    size_t off = 0;
    auto alloc = [&](size_t bytes) -> void* {
        void* p = w + off;
        off += (bytes + 255) & ~(size_t)255;
        return p;
    };
    const size_t MD = (size_t)MT * DD;
    unsigned short* xb    = (unsigned short*)alloc(MD * 2);        // x bf16, reused for conv bf16
    unsigned short* Wyb   = (unsigned short*)alloc((size_t)DD * DD * 2);
    unsigned short* Wxb   = (unsigned short*)alloc((size_t)DD * DD * 2);
    unsigned short* Woutb = (unsigned short*)alloc((size_t)DD * DD * 2);
    unsigned short* gxwb  = (unsigned short*)alloc((size_t)HH * DH * DH * 2);  // [h][j][i]
    unsigned short* gawb  = (unsigned short*)alloc((size_t)HH * DH * DH * 2);  // [h][j][i]
    float* yf  = (float*)alloc(MD * 4);                            // gelu(x@Wy^T)
    float* xpf = (float*)alloc(MD * 4);                            // x_proj -> gate_x -> normed
    float* xcf = (float*)alloc(MD * 4);                            // conv output (f32)
    float* gaf = (float*)alloc(MD * 4);                            // gate_a -> a_scan
    unsigned short* zb = (unsigned short*)alloc(MD * 2);           // (x_lru * y) bf16
    float* Pp    = (float*)alloc((size_t)BB * NC * DD * 4);
    float* Sp    = (float*)alloc((size_t)BB * NC * DD * 4);
    float* carry = (float*)alloc((size_t)BB * NC * DD * 4);

    const int nMD = (int)MD;
    const int cvtBlocks = (nMD + 255) / 256;

    // 1) fp32 -> bf16 conversions (+ gate-weight transpose to [N,K])
    k_f32_to_bf16<<<cvtBlocks, 256, 0, stream>>>(x, xb, nMD);
    k_f32_to_bf16<<<(DD * DD + 255) / 256, 256, 0, stream>>>(Wy, Wyb, DD * DD);
    k_f32_to_bf16<<<(DD * DD + 255) / 256, 256, 0, stream>>>(Wx, Wxb, DD * DD);
    k_f32_to_bf16<<<(DD * DD + 255) / 256, 256, 0, stream>>>(Wout, Woutb, DD * DD);
    k_transpose_gate<<<(HH * DH * DH + 255) / 256, 256, 0, stream>>>(gx_w, gxwb);
    k_transpose_gate<<<(HH * DH * DH + 255) / 256, 256, 0, stream>>>(ga_w, gawb);

    // 2) y = gelu(x @ Wy^T + by); x_proj = x @ Wx^T + bx
    dim3 gemmGrid(DD / 128, MT / 128, 1);
    k_gemm_bf16<1><<<gemmGrid, 256, 0, stream>>>(xb, Wyb, by, yf,
                                                 DD, DD, DD, DD, 0, 0L, 0);
    k_gemm_bf16<0><<<gemmGrid, 256, 0, stream>>>(xb, Wxb, bx, xpf,
                                                 DD, DD, DD, DD, 0, 0L, 0);

    // 3) depthwise causal conv (writes f32 + bf16 into xb)
    k_conv<<<cvtBlocks, 256, 0, stream>>>(xpf, conv_w, conv_b, segpos, xcf, xb);

    // 4) block-diagonal gate GEMMs (per-head batch via grid.z, weights pre-transposed)
    dim3 gateGrid(DH / 128, MT / 128, HH);
    k_gemm_bf16<0><<<gateGrid, 256, 0, stream>>>(xb, gxwb, gx_b, xpf,
                                                 DH, DD, DH, DD,
                                                 DH, (long)DH * DH, DH);
    k_gemm_bf16<0><<<gateGrid, 256, 0, stream>>>(xb, gawb, ga_b, gaf,
                                                 DH, DD, DH, DD,
                                                 DH, (long)DH * DH, DH);

    // 5) gate prep -> normed (xpf), a_scan (gaf)
    k_prep<<<cvtBlocks, 256, 0, stream>>>(xpf, gaf, xcf, a_param, segpos);

    // 6) chunked parallel scan fused with (h * y) -> bf16
    const int nPart = BB * NC * DD;
    k_scan_partial<<<(nPart + 255) / 256, 256, 0, stream>>>(gaf, xpf, Pp, Sp);
    k_scan_carry<<<(BB * DD + 255) / 256, 256, 0, stream>>>(Pp, Sp, carry);
    k_scan_final<<<(nPart + 255) / 256, 256, 0, stream>>>(gaf, xpf, yf, carry, zb);

    // 7) out = (x_lru * y) @ Wout^T + bout
    k_gemm_bf16<0><<<gemmGrid, 256, 0, stream>>>(zb, Woutb, bout, out,
                                                 DD, DD, DD, DD, 0, 0L, 0);
}
